// BesselKANLayer_979252543936
// MI455X (gfx1250) — compile-verified, matrix operand order checked by reading, not run on hardware
//
#include <hip/hip_runtime.h>
#include <hip/hip_bf16.h>

typedef __attribute__((ext_vector_type(16))) _Float16 v16h;
typedef __attribute__((ext_vector_type(8)))  _Float16 v8h;
typedef __attribute__((ext_vector_type(4)))  _Float16 v4h;
typedef __attribute__((ext_vector_type(8)))  float    v8f;
typedef __attribute__((ext_vector_type(4)))  int      v4i;

#define BATCH   8192
#define IDIM    1024
#define ODIM    1024
#define KDIM    (IDIM * 4)      // 4096, k = i*4 + d
#define BM      128
#define BN      128
#define BK      32
#define LDSROW  40              // halves per LDS row (32 data + 8 pad -> 80B stride, 16B aligned)

// ---- gfx1250 async global->LDS path (guarded; falls back to VGPR copy) ----
#if defined(__has_builtin)
#  if __has_builtin(__builtin_amdgcn_global_load_async_to_lds_b128)
#    define USE_ASYNC_LDS 1
#  endif
#endif

#ifdef USE_ASYNC_LDS
// Builtin signature (from clang diagnostic): param0 = v4i* in addrspace(1) (global),
// param1 = v4i* in addrspace(3) (LDS), then imm offset, imm cpol.
typedef v4i __attribute__((address_space(1)))* as1_v4i;
typedef v4i __attribute__((address_space(3)))* as3_v4i;
__device__ __forceinline__ void async_copy16(const _Float16* gsrc, _Float16* ldst) {
    v4i* g = (v4i*)gsrc;              // generic, reinterpret pointee
    v4i* l = (v4i*)ldst;              // generic, reinterpret pointee
    __builtin_amdgcn_global_load_async_to_lds_b128((as1_v4i)g, (as3_v4i)l, 0, 0);
}
#  if __has_builtin(__builtin_amdgcn_s_wait_asynccnt)
#    define WAIT_ASYNC0() __builtin_amdgcn_s_wait_asynccnt(0)
#  else
#    define WAIT_ASYNC0() asm volatile("s_wait_asynccnt 0x0" ::: "memory")
#  endif
#else
#  define WAIT_ASYNC0() ((void)0)
#endif

// ---------------- Kernel 1: T = tanh(x), vectorized float4 ----------------
__global__ void bkan_tanh_kernel(const float* __restrict__ x, float* __restrict__ T) {
    size_t idx = (size_t)blockIdx.x * blockDim.x + threadIdx.x;   // one float4 per thread
    const float4* __restrict__ x4 = (const float4*)x;
    float4* __restrict__ t4 = (float4*)T;
    float4 v = x4[idx];
    v.x = tanhf(v.x); v.y = tanhf(v.y); v.z = tanhf(v.z); v.w = tanhf(v.w);
    t4[idx] = v;
}

// ------- Kernel 2: Wh[o, i*4+d] = (f16) C[i, o, d]  (K-contiguous per column o) -------
__global__ void bkan_packw_kernel(const float* __restrict__ C, _Float16* __restrict__ Wh) {
    int gid = blockIdx.x * blockDim.x + threadIdx.x;  // 1M threads: one (i,o) pair each
    int o = gid >> 10;
    int i = gid & (IDIM - 1);
    float4 c = *(const float4*)(C + (size_t)i * (ODIM * 4) + (size_t)o * 4); // d contiguous
    v4h h = { (_Float16)c.x, (_Float16)c.y, (_Float16)c.z, (_Float16)c.w };
    *(v4h*)(Wh + (size_t)o * KDIM + (size_t)i * 4) = h;                      // d contiguous
}

// ---------------- Kernel 3: WMMA GEMM  y = F(T) @ W ----------------
// Grid: (BATCH/BM = 64, ODIM/BN = 8), block 256 threads = 8 wave32.
// Wave grid inside block: 4 waves along M (32 rows) x 2 waves along N (64 cols).
// Per wave: 2 (M) x 4 (N) accumulators of 16x16 f32 -> 8 wmma per K-step of 32.
// LDS double-buffered: one barrier per K-step; next tile's fill overlaps the WMMAs.
__global__ void __launch_bounds__(256)
bkan_gemm_kernel(const float* __restrict__ T, const _Float16* __restrict__ Wh,
                 float* __restrict__ out) {
    __shared__ __align__(16) _Float16 As[2][BM * LDSROW];   // 2 x 10240 B
    __shared__ __align__(16) _Float16 Bs[2][BN * LDSROW];   // 2 x 10240 B

    const int tid    = threadIdx.x;
    const int lane   = tid & 31;
    const int wave   = tid >> 5;
    const int wave_m = wave >> 1;          // 0..3
    const int wave_n = wave & 1;           // 0..1
    const int wm     = wave_m * 32;
    const int wn     = wave_n * 64;
    const int lmod   = lane & 15;
    const int lhi    = lane >> 4;          // 0 or 1
    const int klo    = lhi * 8;            // K sub-offset per WMMA A/B layout

    const int mBlock = blockIdx.x * BM;
    const int nBlock = blockIdx.y * BN;

    // A-fill addressing: 2 threads per row, each covers 4 consecutive i values.
    const int fa_row  = tid >> 1;          // 0..127
    const int fa_quad = tid & 1;           // 0..1 (i_local quad*4 .. quad*4+3)
    const float* __restrict__ Trow = T + (size_t)(mBlock + fa_row) * IDIM + fa_quad * 4;

    // B-fill addressing: 2 threads per column, each covers 16 halves of K.
    const int fb_col = tid >> 1;           // 0..127
    const int fb_k0  = (tid & 1) * 16;     // 0 or 16
    const _Float16* __restrict__ Wcol =
        Wh + (size_t)(nBlock + fb_col) * KDIM + fb_k0;

    v8f acc[2][4];
    #pragma unroll
    for (int s = 0; s < 2; ++s)
        #pragma unroll
        for (int f = 0; f < 4; ++f)
            acc[s][f] = (v8f){};

    union Frag { v16h v; v8h h[2]; };

    // Issue fills of buffer p for K-tile kt (no waits inside).
    auto fill = [&](int p, int kt) {
        // ---- A tile: basis polynomials of tanh(x), f32 -> f16, into LDS ----
        const float4 t4 = *(const float4*)(Trow + (kt >> 2)); // 4 tanh values
        float tv[4] = { t4.x, t4.y, t4.z, t4.w };
        _Float16 hb[16];
        #pragma unroll
        for (int j = 0; j < 4; ++j) {
            float t  = tv[j];
            float b0 = 1.0f;
            float b1 = t + 1.0f;
            float b2 = 3.0f * t * b1 + b0;   // (2*2-1)*t*b1 + b0
            float b3 = 5.0f * t * b2 + b1;   // (2*3-1)*t*b2 + b1
            hb[j * 4 + 0] = (_Float16)b0;
            hb[j * 4 + 1] = (_Float16)b1;
            hb[j * 4 + 2] = (_Float16)b2;
            hb[j * 4 + 3] = (_Float16)b3;
        }
        _Float16* dstA = &As[p][fa_row * LDSROW + fa_quad * 16];
        *(v8h*)(dstA)     = *(const v8h*)(&hb[0]);   // ds_store_b128
        *(v8h*)(dstA + 8) = *(const v8h*)(&hb[8]);   // ds_store_b128

        // ---- B tile: f16 weights, K-contiguous, into LDS ----
        const _Float16* src = Wcol + kt;
        _Float16* dstB = &Bs[p][fb_col * LDSROW + fb_k0];
#ifdef USE_ASYNC_LDS
        async_copy16(src,     dstB);                  // global_load_async_to_lds_b128
        async_copy16(src + 8, dstB + 8);
#else
        *(v8h*)(dstB)     = *(const v8h*)(src);       // global_load_b128 + ds_store_b128
        *(v8h*)(dstB + 8) = *(const v8h*)(src + 8);
#endif
        if (kt + BK < KDIM)                           // pull next K-tile toward L2/L0
            __builtin_prefetch(src + BK, 0, 0);       // global_prefetch_b8
    };

    fill(0, 0);                                       // prologue
    int p = 0;
    for (int kt = 0; kt < KDIM; kt += BK) {
        WAIT_ASYNC0();        // async fills of buffer p landed in LDS (no-op if sync path)
        __syncthreads();      // emits s_wait_dscnt 0 + barrier: fills done, prior reads done

        if (kt + BK < KDIM)
            fill(p ^ 1, kt + BK);    // overlap next tile's fill with the WMMAs below

        // ---- fragment loads per WMMA 16-bit A/B VGPR layout ----
        Frag af[2], bf[4];
        #pragma unroll
        for (int s = 0; s < 2; ++s) {
            const _Float16* q = &As[p][(wm + s * 16 + lmod) * LDSROW + klo];
            af[s].h[0] = *(const v8h*)(q);        // K 0..7  (lanes 0-15) / 8..15 (16-31)
            af[s].h[1] = *(const v8h*)(q + 16);   // K 16..23 / 24..31
        }
        #pragma unroll
        for (int f = 0; f < 4; ++f) {
            const _Float16* q = &Bs[p][(wn + f * 16 + lmod) * LDSROW + klo];
            bf[f].h[0] = *(const v8h*)(q);
            bf[f].h[1] = *(const v8h*)(q + 16);
        }

        // ---- 8x v_wmma_f32_16x16x32_f16 ----
        #pragma unroll
        for (int s = 0; s < 2; ++s)
            #pragma unroll
            for (int f = 0; f < 4; ++f)
                acc[s][f] = __builtin_amdgcn_wmma_f32_16x16x32_f16(
                    false, af[s].v, false, bf[f].v, (short)0, acc[s][f], false, false);

        p ^= 1;
    }

    // ---- writeback per 16x16 f32 C/D layout: VGPR r -> M = r + 8*lhi, N = lane&15 ----
    #pragma unroll
    for (int s = 0; s < 2; ++s) {
        #pragma unroll
        for (int f = 0; f < 4; ++f) {
            const int rowTop = mBlock + wm + s * 16 + lhi * 8;
            const int col    = nBlock + wn + f * 16 + lmod;
            float* po = out + (size_t)rowTop * ODIM + col;
            #pragma unroll
            for (int r = 0; r < 8; ++r)
                po[(size_t)r * ODIM] = acc[s][f][r];
        }
    }
}

extern "C" void kernel_launch(void* const* d_in, const int* in_sizes, int n_in,
                              void* d_out, int out_size, void* d_ws, size_t ws_size,
                              hipStream_t stream) {
    (void)in_sizes; (void)n_in; (void)out_size; (void)ws_size;
    const float* x = (const float*)d_in[0];                 // [8192, 1024] f32
    const float* C = (const float*)d_in[1];                 // [1024, 1024, 4] f32
    float* out = (float*)d_out;                             // [8192, 1024] f32

    // Workspace: T (tanh, 32 MB f32) | Wh (packed f16 weights, 8 MB)
    float*    T  = (float*)d_ws;
    _Float16* Wh = (_Float16*)((char*)d_ws + (size_t)BATCH * IDIM * sizeof(float));

    // 1) T = tanh(x): 8M elements, float4 per thread
    bkan_tanh_kernel<<<(BATCH * IDIM) / (256 * 4), 256, 0, stream>>>(x, T);

    // 2) Pack/transpose weights to f16 [O, K]: 1M (i,o) pairs
    bkan_packw_kernel<<<(IDIM * ODIM) / 256, 256, 0, stream>>>(C, Wh);

    // 3) WMMA GEMM: 64 x 8 workgroups of 128x128 output tiles
    dim3 grid(BATCH / BM, ODIM / BN);
    bkan_gemm_kernel<<<grid, 256, 0, stream>>>(T, Wh, out);
}
